// PSEADAttention_575525617967
// MI455X (gfx1250) — matrix operand order repaired and checked by reading, not compiled, not run on hardware
//
#include <hip/hip_runtime.h>

// ---------------------------------------------------------------------------
// PSEAD attention for MI455X (gfx1250, wave32, WMMA f32_16x16x32_bf16)
//
// Pipeline:
//   0) cvt_x      : x f32 -> Xb bf16 [16384][1024]
//      cvt_wt x4  : W f32 [k][n] -> Wt bf16 [n][k]
//      cvt_proj   : proj f32 -> Pb (row-major bf16) + Pbt (col-major bf16)
//   1) gemm<2> x2 : Q/K = Xb @ Wt^T + b -> bf16 per-head row-major [B][H][16][64]
//      gemm<0>    : V   = Xb @ Wt^T + b -> bf16 d-major [B][H][64][16]
//   2) attn       : per (b,h):  S0 = Q K^T (once);  per irrep r:
//                     T1 = P_r S0, scores = T1 P_r^T, w = softmax(scores/8),
//                     G += w P_r  (f32 C-registers);  total = G V
//                   (uses scores_r = P (Q K^T) P^T and w(PV) = (wP)V algebra)
//   3) gemm<1>    : out = T @ WtO^T + bo -> f32 [16384][1024]
// ---------------------------------------------------------------------------

typedef unsigned int   u32;
typedef unsigned short u16;
typedef __attribute__((ext_vector_type(16))) __bf16 v16bf;
typedef __attribute__((ext_vector_type(2)))  __bf16 v2bf;
typedef __attribute__((ext_vector_type(2)))  float  v2f;
typedef __attribute__((ext_vector_type(8)))  float  v8f;
typedef __attribute__((ext_vector_type(8)))  u32    v8u;

#define MROWS 16384   // B*K
#define DD    1024
#define NHEAD 16
#define HD    64
#define NIRR  16

// ---- async global->LDS (CDNA5 ASYNCcnt path), guarded fallback -------------
typedef __attribute__((__vector_size__(4 * sizeof(int)))) int v4i_t;

__device__ __forceinline__ void async_copy16(const u16* g, u16* l) {
#if __has_builtin(__builtin_amdgcn_global_load_async_to_lds_b128)
  __builtin_amdgcn_global_load_async_to_lds_b128((v4i_t*)g, (v4i_t*)l, 0, 0);
#else
  *(uint4*)l = *(const uint4*)g;
#endif
}
__device__ __forceinline__ void async_wait() {
#if __has_builtin(__builtin_amdgcn_s_wait_asynccnt)
  __builtin_amdgcn_s_wait_asynccnt(0);
#endif
}

// ---- native bf16 converts (backend emits packed v_cvt ops) ------------------
__device__ __forceinline__ u32 pack_bf16(float lo, float hi) {
  v2f f;
  f[0] = lo;
  f[1] = hi;
  const v2bf h = __builtin_convertvector(f, v2bf);
  return __builtin_bit_cast(u32, h);
}
__device__ __forceinline__ u16 f2bf(float x) {
  const __bf16 h = (__bf16)x;
  return __builtin_bit_cast(u16, h);
}
__device__ __forceinline__ v8f wmma_bf16(v8u a, v8u b, v8f c) {
  return __builtin_amdgcn_wmma_f32_16x16x32_bf16(
      false, __builtin_bit_cast(v16bf, a),
      false, __builtin_bit_cast(v16bf, b),
      (short)0, c, false, false);
}

// ---------------------------------------------------------------------------
__global__ __launch_bounds__(256) void cvt_x_kernel(const float* __restrict__ X,
                                                    u16* __restrict__ Xb) {
  const size_t i = ((size_t)blockIdx.x * 256 + threadIdx.x) * 4;
  const float4 f = *(const float4*)(X + i);
  uint2 p;
  p.x = pack_bf16(f.x, f.y);
  p.y = pack_bf16(f.z, f.w);
  *(uint2*)(Xb + i) = p;
}

__global__ __launch_bounds__(256) void cvt_wt_kernel(const float* __restrict__ W,
                                                     u16* __restrict__ Wt) {
  __shared__ float tile[32][33];
  const int x = threadIdx.x & 31;
  const int y = threadIdx.x >> 5;
  const int k0 = blockIdx.x * 32;
  const int n0 = blockIdx.y * 32;
  #pragma unroll
  for (int i = 0; i < 4; ++i)
    tile[y + 8 * i][x] = W[(size_t)(k0 + y + 8 * i) * DD + n0 + x];
  __syncthreads();
  #pragma unroll
  for (int i = 0; i < 4; ++i)
    Wt[(size_t)(n0 + y + 8 * i) * DD + k0 + x] = f2bf(tile[x][y + 8 * i]);
}

// proj f32 [16][16][16] -> Pb[r][i][m] bf16 and Pbt[r][n][j] = P[j][n] bf16
__global__ __launch_bounds__(256) void cvt_proj_kernel(const float* __restrict__ proj,
                                                       u16* __restrict__ Pb,
                                                       u16* __restrict__ Pbt) {
  const int t = blockIdx.x * 256 + threadIdx.x;   // 0..4095
  const int r = t >> 8, i = (t >> 4) & 15, m = t & 15;
  const u16 v = f2bf(proj[t]);
  Pb[t] = v;
  Pbt[r * 256 + m * 16 + i] = v;
}

// ---------------------------------------------------------------------------
// GEMM: C[M x 1024] = A[M x 1024] @ Wt[1024 x 1024]^T + bias  (A, Wt bf16)
// Block 128x64, BK=32, 256 threads (8 waves); wave -> 16 rows x 64 cols.
// MODE 0: bf16 d-major [B][H][64][16];  MODE 1: f32 row-major [M][1024];
// MODE 2: bf16 per-head row-major [B][H][16][64]
// ---------------------------------------------------------------------------
template <int MODE>
__global__ __launch_bounds__(256) void gemm_kernel(const u16* __restrict__ Ab,
                                                   const u16* __restrict__ Wt,
                                                   const float* __restrict__ bias,
                                                   void* __restrict__ Out) {
  __shared__ __align__(16) u16 smemA[128 * 40];
  __shared__ __align__(16) u16 smemB[64 * 40];

  const int tid  = threadIdx.x;
  const int lane = tid & 31;
  const int wave = tid >> 5;
  const int half = lane >> 4;
  const int l16  = lane & 15;
  const int bm   = blockIdx.x * 128;
  const int bn   = blockIdx.y * 64;

  v8f cacc[4];
  #pragma unroll
  for (int c = 0; c < 4; ++c) cacc[c] = v8f{0.f,0.f,0.f,0.f,0.f,0.f,0.f,0.f};

  for (int k0 = 0; k0 < DD; k0 += 32) {
    #pragma unroll
    for (int i = 0; i < 2; ++i) {
      const int cid = tid + 256 * i;
      const int row = cid >> 2, ch = cid & 3;
      async_copy16(Ab + (size_t)(bm + row) * DD + k0 + ch * 8,
                   &smemA[row * 40 + ch * 8]);
    }
    {
      const int row = tid >> 2, ch = tid & 3;
      async_copy16(Wt + (size_t)(bn + row) * DD + k0 + ch * 8,
                   &smemB[row * 40 + ch * 8]);
    }
    async_wait();
    __syncthreads();

    const u32* Aw = (const u32*)smemA;
    const u32* Bw = (const u32*)smemB;
    v8u a;
    const int arow = wave * 16 + l16;
    #pragma unroll
    for (int v = 0; v < 8; ++v)
      a[v] = Aw[arow * 20 + (v >> 2) * 8 + half * 4 + (v & 3)];
    #pragma unroll
    for (int c = 0; c < 4; ++c) {
      v8u b;
      #pragma unroll
      for (int v = 0; v < 8; ++v)
        b[v] = Bw[(c * 16 + l16) * 20 + (v >> 2) * 8 + half * 4 + (v & 3)];
      cacc[c] = wmma_bf16(a, b, cacc[c]);
    }
    __syncthreads();
  }

  const int rowbase = bm + wave * 16;   // multiple of 16
  #pragma unroll
  for (int c = 0; c < 4; ++c) {
    const int n = bn + c * 16 + l16;
    const float bs = bias[n];
    const int bidx = rowbase >> 4;
    const int h = n >> 6, d = n & 63;
    if (MODE == 0) {
      uint4 pk;
      pk.x = pack_bf16(cacc[c][0] + bs, cacc[c][1] + bs);
      pk.y = pack_bf16(cacc[c][2] + bs, cacc[c][3] + bs);
      pk.z = pack_bf16(cacc[c][4] + bs, cacc[c][5] + bs);
      pk.w = pack_bf16(cacc[c][6] + bs, cacc[c][7] + bs);
      u16* O = (u16*)Out;
      *(uint4*)(O + ((((size_t)bidx * NHEAD + h) * HD + d) * 16 + 8 * half)) = pk;
    } else if (MODE == 2) {
      u16* O = (u16*)Out;
      #pragma unroll
      for (int r = 0; r < 8; ++r) {
        const int k = r + 8 * half;
        O[(((size_t)bidx * NHEAD + h) * 16 + k) * HD + d] = f2bf(cacc[c][r] + bs);
      }
    } else {
      float* O = (float*)Out;
      #pragma unroll
      for (int r = 0; r < 8; ++r)
        O[(size_t)(rowbase + r + 8 * half) * DD + n] = cacc[c][r] + bs;
    }
  }
}

// ---------------------------------------------------------------------------
// Attention: one wave per (b,h).
//   S0 = Q K^T        (2 WMMAs, once)
//   per irrep: T1 = P S0 (1), scores = T1 P^T (1), softmax, G += w P (1)
//   total = G V       (4 WMMAs)
// All 16->32 K-padding comes from zeroed LDS slots (no cndmasks).
// P tiles arrive per irrep via global_load_async_to_lds (pre-converted bf16).
// ---------------------------------------------------------------------------
__global__ __launch_bounds__(32) void attn_kernel(const u16* __restrict__ Qg,
                                                  const u16* __restrict__ Kg,
                                                  const u16* __restrict__ Vg,
                                                  const u16* __restrict__ Pb,
                                                  const u16* __restrict__ Pbt,
                                                  u16* __restrict__ T) {
  const int bh   = blockIdx.x;
  const int lane = threadIdx.x & 31;
  const int half = lane >> 4;
  const int l16  = lane & 15;

  __shared__ __align__(16) u16 Qs[16 * 80], Ks[16 * 80];  // k-major [i][64d]
  __shared__ __align__(16) u16 Vt[64 * 40];               // d-major, upper K zero
  __shared__ __align__(16) u16 S0t[16 * 40];              // S0 col-major [col][j]
  __shared__ __align__(16) u16 T1s[16 * 40];              // T1 row-major [i][n]
  __shared__ __align__(16) u16 Gs[16 * 40];               // G row-major [i][m]
  __shared__ __align__(16) u16 Pt[16 * 40];               // P row-major bf16
  __shared__ __align__(16) u16 Ptt[16 * 40];              // P col-major bf16
  __shared__ __align__(16) u16 wLds[16][32];
  __shared__ float sLds[16][17];
  __shared__ u16 Ts[16][72];

  // ---- zero the padded upper K-slots (one-time) ----------------------------
  {
    const uint4 z4 = {0u, 0u, 0u, 0u};
    #pragma unroll
    for (int i = 0; i < 4; ++i) {
      const int cid = lane + 32 * i;
      const int row = cid >> 1, hr = cid & 1;
      *(uint4*)&Vt[row * 40 + 16 + hr * 8] = z4;
    }
    {
      const int row = lane >> 1, hr = lane & 1;
      const int o = row * 40 + 16 + hr * 8;
      *(uint4*)&S0t[o] = z4;  *(uint4*)&T1s[o] = z4;  *(uint4*)&Gs[o] = z4;
      *(uint4*)&Pt[o]  = z4;  *(uint4*)&Ptt[o] = z4;
    }
    if (lane < 16) {
      *(uint4*)&wLds[lane][16] = z4;
      *(uint4*)&wLds[lane][24] = z4;
    }
  }

  // ---- async-load Q,K (row-major 2KB) and V (d-major 2KB) ------------------
  {
    const u16* qb = Qg + (size_t)bh * 1024;
    const u16* kb = Kg + (size_t)bh * 1024;
    const u16* vb = Vg + (size_t)bh * 1024;
    #pragma unroll
    for (int i = 0; i < 4; ++i) {
      const int pI = lane + 32 * i;               // 0..127 chunks of 16B
      {
        const int row = pI >> 3, ch = pI & 7;     // 8 chunks per 128B row
        async_copy16(qb + pI * 8, &Qs[row * 80 + ch * 8]);
        async_copy16(kb + pI * 8, &Ks[row * 80 + ch * 8]);
      }
      {
        const int row = pI >> 1, hr = pI & 1;     // 2 chunks per 32B row
        async_copy16(vb + pI * 8, &Vt[row * 40 + hr * 8]);
      }
    }
  }
  async_wait();
  __syncthreads();

  const v8f zero = {0.f, 0.f, 0.f, 0.f, 0.f, 0.f, 0.f, 0.f};

  // ---- S0 = Q @ K^T (16x16, K = 64) ----------------------------------------
  v8f cs0 = zero;
  {
    const u32* Qw = (const u32*)Qs;
    const u32* Kw = (const u32*)Ks;
    #pragma unroll
    for (int ch = 0; ch < 2; ++ch) {
      v8u aq, bk;
      #pragma unroll
      for (int v = 0; v < 8; ++v) {
        aq[v] = Qw[l16 * 40 + ch * 16 + (v >> 2) * 8 + half * 4 + (v & 3)];
        bk[v] = Kw[l16 * 40 + ch * 16 + (v >> 2) * 8 + half * 4 + (v & 3)];
      }
      cs0 = wmma_bf16(aq, bk, cs0);
    }
  }
  // store S0 col-major: lane holds col=l16, rows rr+8*half (contiguous)
  {
    u32* sw = (u32*)&S0t[l16 * 40 + 8 * half];
    sw[0] = pack_bf16(cs0[0], cs0[1]);  sw[1] = pack_bf16(cs0[2], cs0[3]);
    sw[2] = pack_bf16(cs0[4], cs0[5]);  sw[3] = pack_bf16(cs0[6], cs0[7]);
  }
  __syncthreads();

  v8f G = zero;   // G = sum_r w_r @ P_r, accumulated in C registers

  const int prow = lane >> 1, phr = lane & 1;   // P tile staging coords

  for (int r = 0; r < NIRR; ++r) {
    // ---- async-stage P_r (row-major + col-major, pre-converted bf16) ------
    async_copy16(Pb  + r * 256 + prow * 16 + phr * 8, &Pt[prow * 40 + phr * 8]);
    async_copy16(Pbt + r * 256 + prow * 16 + phr * 8, &Ptt[prow * 40 + phr * 8]);
    async_wait();
    __syncthreads();

    const u32* Ptw  = (const u32*)Pt;
    const u32* Pttw = (const u32*)Ptt;
    const u32* S0w  = (const u32*)S0t;
    const u32* T1w  = (const u32*)T1s;

    // ---- T1 = P @ S0 ------------------------------------------------------
    v8u pf, bs0;
    #pragma unroll
    for (int v = 0; v < 8; ++v) {
      pf[v]  = Ptw[l16 * 20 + (v >> 2) * 8 + half * 4 + (v & 3)];
      bs0[v] = S0w[l16 * 20 + (v >> 2) * 8 + half * 4 + (v & 3)];
    }
    v8f t1 = wmma_bf16(pf, bs0, zero);

    // scatter T1 -> row-major T1s
    #pragma unroll
    for (int rr = 0; rr < 8; ++rr)
      T1s[(rr + 8 * half) * 40 + l16] = f2bf(t1[rr]);
    __syncthreads();

    // ---- scores = T1 @ P^T ------------------------------------------------
    v8u a1, bpt;
    #pragma unroll
    for (int v = 0; v < 8; ++v) {
      a1[v]  = T1w[l16 * 20 + (v >> 2) * 8 + half * 4 + (v & 3)];
      bpt[v] = Ptw[l16 * 20 + (v >> 2) * 8 + half * 4 + (v & 3)];   // B[k=m][n] = P[n][m] -> Pt rows
    }
    v8f sc = wmma_bf16(a1, bpt, zero);

    // ---- row softmax (scale 1/sqrt(64) = 0.125) ---------------------------
    #pragma unroll
    for (int rr = 0; rr < 8; ++rr)
      sLds[rr + 8 * half][l16] = sc[rr];
    __syncthreads();
    if (lane < 16) {
      float mx = sLds[lane][0];
      #pragma unroll
      for (int j = 1; j < 16; ++j) mx = fmaxf(mx, sLds[lane][j]);
      float sum = 0.f;
      #pragma unroll
      for (int j = 0; j < 16; ++j) {
        const float e = __expf((sLds[lane][j] - mx) * 0.125f);
        sLds[lane][j] = e;
        sum += e;
      }
      const float inv = 1.f / sum;
      #pragma unroll
      for (int j = 0; j < 16; ++j)
        wLds[lane][j] = f2bf(sLds[lane][j] * inv);
    }
    __syncthreads();

    // ---- G += w @ P  (B-frag from col-major Ptt) --------------------------
    v8u wa, bp;
    #pragma unroll
    for (int v = 0; v < 8; ++v) {
      wa[v] = *(const u32*)&wLds[l16][(v >> 2) * 16 + half * 8 + (v & 3) * 2];
      bp[v] = Pttw[l16 * 20 + (v >> 2) * 8 + half * 4 + (v & 3)];
    }
    G = wmma_bf16(wa, bp, G);
    __syncthreads();   // Pt/Ptt/T1s rewritten next irrep
  }

  // ---- total = G @ V --------------------------------------------------------
  #pragma unroll
  for (int rr = 0; rr < 8; ++rr)
    Gs[(rr + 8 * half) * 40 + l16] = f2bf(G[rr]);
  __syncthreads();
  v8u ga;
  {
    const u32* Gw = (const u32*)Gs;
    #pragma unroll
    for (int v = 0; v < 8; ++v)
      ga[v] = Gw[l16 * 20 + (v >> 2) * 8 + half * 4 + (v & 3)];
  }
  const u32* Vw = (const u32*)Vt;
  #pragma unroll
  for (int c = 0; c < 4; ++c) {
    const int col = c * 16 + l16;
    v8u bv;
    #pragma unroll
    for (int v = 0; v < 8; ++v)
      bv[v] = Vw[col * 20 + (v >> 2) * 8 + half * 4 + (v & 3)];
    v8f o = wmma_bf16(ga, bv, zero);
    #pragma unroll
    for (int rr = 0; rr < 8; ++rr)
      Ts[rr + 8 * half][col] = f2bf(o[rr]);
  }
  __syncthreads();
  const int bidx = bh >> 4, h = bh & 15;
  u16* To = T + (size_t)bidx * 16 * DD + h * HD;
  for (int pI = lane; pI < 512; pI += 32) {
    const int row = pI >> 5, cp = pI & 31;
    *(u32*)(To + (size_t)row * DD + 2 * cp) = *(const u32*)&Ts[row][2 * cp];
  }
}

// ---------------------------------------------------------------------------
extern "C" void kernel_launch(void* const* d_in, const int* in_sizes, int n_in,
                              void* d_out, int out_size, void* d_ws, size_t ws_size,
                              hipStream_t stream) {
  const float* x    = (const float*)d_in[0];
  const float* proj = (const float*)d_in[1];
  const float* Wq   = (const float*)d_in[2];
  const float* bq   = (const float*)d_in[3];
  const float* Wk   = (const float*)d_in[4];
  const float* bk   = (const float*)d_in[5];
  const float* Wv   = (const float*)d_in[6];
  const float* bv   = (const float*)d_in[7];
  const float* Wo   = (const float*)d_in[8];
  const float* bo   = (const float*)d_in[9];

  u16* Xb  = (u16*)d_ws;                                 // 32MB
  u16* WtQ = Xb  + (size_t)MROWS * DD;                   // 2MB each
  u16* WtK = WtQ + (size_t)DD * DD;
  u16* WtV = WtK + (size_t)DD * DD;
  u16* WtO = WtV + (size_t)DD * DD;
  u16* Qws = WtO + (size_t)DD * DD;                      // 32MB each
  u16* Kws = Qws + (size_t)MROWS * DD;
  u16* Vws = Kws + (size_t)MROWS * DD;
  u16* Tws = Vws + (size_t)MROWS * DD;                   // 32MB
  u16* Pbw = Tws + (size_t)MROWS * DD;                   // 8KB
  u16* Pbtw = Pbw + (size_t)NIRR * 256;                  // 8KB

  cvt_x_kernel<<<dim3(MROWS * DD / 1024), dim3(256), 0, stream>>>(x, Xb);
  dim3 tgrid(32, 32), tblk(256);
  cvt_wt_kernel<<<tgrid, tblk, 0, stream>>>(Wq, WtQ);
  cvt_wt_kernel<<<tgrid, tblk, 0, stream>>>(Wk, WtK);
  cvt_wt_kernel<<<tgrid, tblk, 0, stream>>>(Wv, WtV);
  cvt_wt_kernel<<<tgrid, tblk, 0, stream>>>(Wo, WtO);
  cvt_proj_kernel<<<dim3(16), dim3(256), 0, stream>>>(proj, Pbw, Pbtw);

  dim3 ggrid(MROWS / 128, DD / 64), gblk(256);
  gemm_kernel<2><<<ggrid, gblk, 0, stream>>>(Xb, WtQ, bq, Qws);
  gemm_kernel<2><<<ggrid, gblk, 0, stream>>>(Xb, WtK, bk, Kws);
  gemm_kernel<0><<<ggrid, gblk, 0, stream>>>(Xb, WtV, bv, Vws);
  attn_kernel<<<dim3(MROWS), dim3(32), 0, stream>>>(Qws, Kws, Vws, Pbw, Pbtw, Tws);
  gemm_kernel<1><<<ggrid, gblk, 0, stream>>>(Tws, WtO, bo, d_out);
}
